// SparseRPNNet4_34127810134190
// MI455X (gfx1250) — matrix operand order repaired and checked by reading, not compiled, not run on hardware
//
#include <hip/hip_runtime.h>
#include <stdint.h>

#define NPTS   300000
#define KOFF   27
#define C_IN   256
#define C_MID  128
#define C_S    32
#define C_OUT  8
#define BN_EPS 1e-5f

#define TM 128                               // points per workgroup in conv
#define NWG1 ((NPTS + TM - 1) / TM)          // 2344
#define PTS2 1024
#define NWG2 ((NPTS + PTS2 - 1) / PTS2)      // 293
#define W1B_ELEMS (KOFF * 8 * 8 * 512)       // 884736 packed bf16 elements
#define TOTCH (KOFF * 8)                     // 216 k-chunks of 32 input channels

typedef __attribute__((ext_vector_type(16))) __bf16 v16bf;
typedef __attribute__((ext_vector_type(8)))  float  v8f;

union Frag   { uint4 u[2]; v16bf v; };
union Pack16 { __bf16 h[16]; uint4 u[2]; };

// ---------------------------------------------------------------------------
// Pack W1 [27][256][128] f32 -> bf16 fragments in WMMA B-matrix lane layout.
// Layout index: (((k*8 + kc)*8 + ct)*512 + lane*16 + e)
//   B (32x16 bf16): lanes 0-15 hold K=0-7,16-23, lanes 16-31 hold K=8-15,24-31; col = ct*16+(lane&15)
// ---------------------------------------------------------------------------
__global__ __launch_bounds__(256)
void k_pack_w1(const float* __restrict__ W1, __bf16* __restrict__ W1b) {
  const int g = blockIdx.x * 256 + threadIdx.x;
  if (g >= W1B_ELEMS) return;
  const int e    = g & 15;
  const int lane = (g >> 4) & 31;
  const int ct   = (g >> 9) & 7;
  const int kc   = (g >> 12) & 7;
  const int k    = g >> 15;
  const int Krow = ((e >> 3) << 4) + ((lane >> 4) << 3) + (e & 7);     // 0..31 within the 32-wide chunk
  const int row  = kc * 32 + Krow;             // input channel 0..255
  const int col  = ct * 16 + (lane & 15);      // output channel 0..127
  W1b[g] = (__bf16)W1[((size_t)k * C_IN + row) * C_MID + col];
}

// ---------------------------------------------------------------------------
// Main sparse conv: h1[n, 0..127] = sum_k mask * x[idx[n,k]] @ W1[k]
// WG = 256 threads (8 waves), 128 points x 128 output channels per WG.
// bf16 WMMA 16x16x32, f32 accumulate.
// 3-stage software pipeline over double-buffered LDS:
//   iter t:  wait async / barrier
//            cvt+store A(t+1) to LDS, issue async B DMA(t+1)   (fv in flight)
//            issue gather loads for A(t+2) into registers      (no wait)
//            compute chunk t (8 x WMMA)
// so the random-gather HBM latency and the B DMA both overlap the WMMAs.
// ---------------------------------------------------------------------------
__global__ __launch_bounds__(256)
void k_conv(const float* __restrict__ x, const int* __restrict__ nbr_idx,
            const unsigned char* __restrict__ nbr_mask,
            const __bf16* __restrict__ W1b,
            float* __restrict__ h1, float* __restrict__ part1) {
  // buffer b at smem + b*16384:  [A: 128x32 bf16 = 8KB][B: 8x512 bf16 = 8KB]
  __shared__ __align__(16) unsigned char smem[32768];
  float* red = (float*)smem;              // reused for stats reduction (16 KB)

  const int tid  = threadIdx.x;
  const int wave = tid >> 5;
  const int lane = tid & 31;
  const int n0   = blockIdx.x * TM;

  v8f acc[8] = {};

  // A-staging role: thread stages 16 channels of one point
  const int  ar    = tid >> 1;          // row 0..127 of the tile
  const int  ah    = tid & 1;           // which 16-ch half of the 32-chunk
  const int  an    = n0 + ar;
  const bool rowok = (an < NPTS);

  const unsigned lds_base = (unsigned)(size_t)smem;     // low 32 bits = LDS byte addr

  // per-wave A fragment element offset within an A buffer:
  // A (16x32 bf16): lanes 0-15 -> K base 0, lanes 16-31 -> K base 8
  const int aoff_e = (wave * 16 + (lane & 15)) * 32 + ((lane >> 4) * 8);

  // pipeline state
  bool  m = false;        // gather mask for current k (loop-carried)
  int   idx = 0;          // gather row for current k
  float fv[16];           // gathered f32 data for the chunk currently in flight

  // ---- issue gather loads for chunk t into fv (no explicit wait here) ----
  auto gatherIssue = [&](int t) {
    if (t >= TOTCH) {
#pragma unroll
      for (int j = 0; j < 16; ++j) fv[j] = 0.f;
      return;
    }
    const int k  = t >> 3;
    const int kc = t & 7;
    if (kc == 0) {
      m = false; idx = 0;
      if (rowok) {
        m = (nbr_mask[(size_t)an * KOFF + k] != 0);
        if (m) idx = nbr_idx[(size_t)an * KOFF + k];
      }
      if (m) {
        // warm the remainder of this point's 1KB row (consumed over next 7 chunks)
        const char* row = (const char*)(x + (size_t)idx * C_IN);
        __builtin_prefetch(row + 256, 0, 3);
        __builtin_prefetch(row + 512, 0, 3);
        __builtin_prefetch(row + 768, 0, 3);
      }
    }
    if (m) {
      const float* xr = x + (size_t)idx * C_IN + kc * 32 + ah * 16;
      const float4 f0 = *(const float4*)(xr + 0);
      const float4 f1 = *(const float4*)(xr + 4);
      const float4 f2 = *(const float4*)(xr + 8);
      const float4 f3 = *(const float4*)(xr + 12);
      fv[0]=f0.x; fv[1]=f0.y; fv[2]=f0.z; fv[3]=f0.w;
      fv[4]=f1.x; fv[5]=f1.y; fv[6]=f1.z; fv[7]=f1.w;
      fv[8]=f2.x; fv[9]=f2.y; fv[10]=f2.z; fv[11]=f2.w;
      fv[12]=f3.x; fv[13]=f3.y; fv[14]=f3.z; fv[15]=f3.w;
    } else {
#pragma unroll
      for (int j = 0; j < 16; ++j) fv[j] = 0.f;
    }
  };

  // ---- convert fv -> bf16, store to LDS buffer t&1, issue async B DMA(t) ----
  auto cvtStore = [&](int t) {
    if (t >= TOTCH) return;
    const unsigned bufb = (t & 1) ? 16384u : 0u;
    Pack16 pk;
#pragma unroll
    for (int j = 0; j < 16; ++j) pk.h[j] = (__bf16)fv[j];
    uint4* adst = (uint4*)(smem + bufb + (unsigned)(ar * 32 + ah * 16) * 2u);
    adst[0] = pk.u[0];
    adst[1] = pk.u[1];

    // B: async DMA 8 KB of packed W1 fragments (L2-resident) into LDS
    const __bf16* gb    = W1b + (size_t)t * 4096;        // uniform (SGPR base)
    const unsigned voff = (unsigned)tid * 32u;           // per-lane byte offset
    const unsigned ldsB = lds_base + bufb + 8192u + (unsigned)tid * 32u;
    asm volatile("global_load_async_to_lds_b128 %0, %1, %2 offset:0"
                 :: "v"(ldsB), "v"(voff), "s"(gb) : "memory");
    asm volatile("global_load_async_to_lds_b128 %0, %1, %2 offset:16"
                 :: "v"(ldsB), "v"(voff), "s"(gb) : "memory");
  };

  // pipeline prologue
  gatherIssue(0);
  cvtStore(0);
  gatherIssue(1);

  for (int t = 0; t < TOTCH; ++t) {
    asm volatile("s_wait_asynccnt 0x0" ::: "memory");   // our B DMA for chunk t done
    __syncthreads();                                    // everyone's staging of t visible

    cvtStore(t + 1);                                    // fv(t+1) -> LDS, async B(t+1)
    gatherIssue(t + 2);                                 // fire HBM gather for t+2

    // ---- compute chunk t: 1 A fragment x 8 B fragments per wave ----
    const unsigned bufb = (t & 1) ? 16384u : 0u;
    const __bf16* sAb = (const __bf16*)(smem + bufb);
    const __bf16* sBb = (const __bf16*)(smem + bufb + 8192u);

    Frag a;
    a.u[0] = *(const uint4*)(sAb + aoff_e);
    a.u[1] = *(const uint4*)(sAb + aoff_e + 16);

    Frag b[2];
    {
      const __bf16* bp = sBb + lane * 16;
      b[0].u[0] = *(const uint4*)bp;
      b[0].u[1] = *(const uint4*)(bp + 8);
    }
#pragma unroll
    for (int ct = 0; ct < 8; ++ct) {
      if (ct < 7) {
        const __bf16* bp = sBb + ((ct + 1) * 512 + lane * 16);
        b[(ct + 1) & 1].u[0] = *(const uint4*)bp;
        b[(ct + 1) & 1].u[1] = *(const uint4*)(bp + 8);
      }
      acc[ct] = __builtin_amdgcn_wmma_f32_16x16x32_bf16(
          false, a.v, false, b[ct & 1].v, (short)0, acc[ct], false, false);
    }
  }

  // ---- epilogue: store h1 tile (C layout: VGPR v, lanes0-15 M=v, 16-31 M=v+8) ----
  const int col  = lane & 15;
  const int radd = (lane >> 4) * 8;
#pragma unroll
  for (int ct = 0; ct < 8; ++ct) {
#pragma unroll
    for (int v = 0; v < 8; ++v) {
      const int n = n0 + wave * 16 + radd + v;
      if (n < NPTS) h1[(size_t)n * C_MID + ct * 16 + col] = acc[ct][v];
    }
  }

  // ---- deterministic per-channel partial sum / sumsq for BN1 ----
  __syncthreads();
#pragma unroll
  for (int ct = 0; ct < 8; ++ct) {
    float s = 0.f, q = 0.f;
#pragma unroll
    for (int v = 0; v < 8; ++v) { const float t = acc[ct][v]; s += t; q += t * t; }
    red[tid * 16 + ct * 2 + 0] = s;
    red[tid * 16 + ct * 2 + 1] = q;
  }
  __syncthreads();
  if (tid < C_MID) {
    const int ct2 = tid >> 4, c2 = tid & 15;
    float s = 0.f, q = 0.f;
    for (int w = 0; w < 8; ++w) {
      const int t0 = w * 32 + c2, t1 = t0 + 16;
      s += red[t0 * 16 + ct2 * 2]     + red[t1 * 16 + ct2 * 2];
      q += red[t0 * 16 + ct2 * 2 + 1] + red[t1 * 16 + ct2 * 2 + 1];
    }
    part1[(size_t)blockIdx.x * 256 + tid]       = s;
    part1[(size_t)blockIdx.x * 256 + 128 + tid] = q;
  }
}

// ---------------------------------------------------------------------------
// Finalize BN1: stats[c] = gamma*rstd, stats[128+c] = beta - mean*gamma*rstd
// ---------------------------------------------------------------------------
__global__ __launch_bounds__(256)
void k_reduce1(const float* __restrict__ part1, const float* __restrict__ gamma,
               const float* __restrict__ beta, float* __restrict__ stats) {
  __shared__ float buf[256];
  const int tid = threadIdx.x;
  float a = 0.f;
  for (int wg = 0; wg < NWG1; ++wg) a += part1[(size_t)wg * 256 + tid];
  buf[tid] = a;
  __syncthreads();
  if (tid < C_MID) {
    const float invn = 1.0f / (float)NPTS;
    const float mean = buf[tid] * invn;
    float var = buf[128 + tid] * invn - mean * mean;
    var = var < 0.f ? 0.f : var;
    const float sc = gamma[tid] * rsqrtf(var + BN_EPS);
    stats[tid]       = sc;
    stats[128 + tid] = beta[tid] - mean * sc;
  }
}

// ---------------------------------------------------------------------------
// h2 = relu(BN1(h1)) @ W2, with fused BN2 stats partials. 1024 pts per WG.
// ---------------------------------------------------------------------------
__global__ __launch_bounds__(256)
void k_stage2(const float* __restrict__ h1, const float* __restrict__ stats1,
              const float* __restrict__ W2, float* __restrict__ h2,
              float* __restrict__ part2) {
  __shared__ float sW2[C_MID * C_S];             // 16 KB
  __shared__ float sSc[C_MID], sSh[C_MID];
  __shared__ __align__(16) float sX[32 * C_MID]; // 16 KB
  __shared__ float red2[256 * 8];                // 8 KB

  const int tid = threadIdx.x;
  for (int i = tid; i < C_MID * C_S; i += 256) sW2[i] = W2[i];
  if (tid < C_MID) { sSc[tid] = stats1[tid]; sSh[tid] = stats1[128 + tid]; }

  const int prow = tid >> 3;
  const int cgrp = (tid & 7) * 4;
  float ts[4] = {0,0,0,0}, tq[4] = {0,0,0,0};

  const int flat = tid * 16;
  const int lr   = flat >> 7;
  const int lc0  = flat & 127;

  for (int it = 0; it < 32; ++it) {
    const int p0 = blockIdx.x * PTS2 + it * 32;
    __syncthreads();
    // load 32x128 block of h1 with BN + ReLU applied
    {
      const int n = p0 + lr;
      float v[16];
      if (n < NPTS) {
        const float4* src = (const float4*)(h1 + (size_t)n * C_MID + lc0);
#pragma unroll
        for (int j = 0; j < 4; ++j) {
          const float4 f = src[j];
          v[4*j]=f.x; v[4*j+1]=f.y; v[4*j+2]=f.z; v[4*j+3]=f.w;
        }
#pragma unroll
        for (int j = 0; j < 16; ++j) {
          const int c = lc0 + j;
          const float t = v[j] * sSc[c] + sSh[c];
          v[j] = t > 0.f ? t : 0.f;
        }
      } else {
#pragma unroll
        for (int j = 0; j < 16; ++j) v[j] = 0.f;
      }
#pragma unroll
      for (int j = 0; j < 16; ++j) sX[flat + j] = v[j];
    }
    __syncthreads();

    const int n2 = p0 + prow;
    float o[4] = {0,0,0,0};
    const float* xr = sX + prow * C_MID;
#pragma unroll 4
    for (int mm = 0; mm < C_MID; ++mm) {
      const float a = xr[mm];
#pragma unroll
      for (int j = 0; j < 4; ++j) o[j] += a * sW2[mm * C_S + cgrp + j];
    }
    if (n2 < NPTS) {
#pragma unroll
      for (int j = 0; j < 4; ++j) {
        h2[(size_t)n2 * C_S + cgrp + j] = o[j];
        ts[j] += o[j];
        tq[j] += o[j] * o[j];
      }
    }
  }

  __syncthreads();
#pragma unroll
  for (int j = 0; j < 4; ++j) { red2[tid*8+j] = ts[j]; red2[tid*8+4+j] = tq[j]; }
  __syncthreads();
  if (tid < C_S) {
    const int cg = tid >> 2, j = tid & 3;
    float s = 0.f, q = 0.f;
    for (int pt = 0; pt < 32; ++pt) {
      const int t = pt * 8 + cg;
      s += red2[t*8+j];
      q += red2[t*8+4+j];
    }
    part2[(size_t)blockIdx.x * 64 + tid]      = s;
    part2[(size_t)blockIdx.x * 64 + 32 + tid] = q;
  }
}

__global__ __launch_bounds__(64)
void k_reduce2(const float* __restrict__ part2, const float* __restrict__ gamma,
               const float* __restrict__ beta, float* __restrict__ stats) {
  __shared__ float buf[64];
  const int tid = threadIdx.x;
  float a = 0.f;
  for (int wg = 0; wg < NWG2; ++wg) a += part2[(size_t)wg * 64 + tid];
  buf[tid] = a;
  __syncthreads();
  if (tid < C_S) {
    const float invn = 1.0f / (float)NPTS;
    const float mean = buf[tid] * invn;
    float var = buf[32 + tid] * invn - mean * mean;
    var = var < 0.f ? 0.f : var;
    const float sc = gamma[tid] * rsqrtf(var + BN_EPS);
    stats[tid]      = sc;
    stats[32 + tid] = beta[tid] - mean * sc;
  }
}

// ---------------------------------------------------------------------------
// out = BN2(h2) @ W3   (one point per thread)
// ---------------------------------------------------------------------------
__global__ __launch_bounds__(256)
void k_stage3(const float* __restrict__ h2, const float* __restrict__ stats2,
              const float* __restrict__ W3, float* __restrict__ out) {
  __shared__ float sW3[C_S * C_OUT];
  __shared__ float sSc[C_S], sSh[C_S];
  const int tid = threadIdx.x;
  if (tid < C_S * C_OUT) sW3[tid] = W3[tid];
  if (tid < C_S) { sSc[tid] = stats2[tid]; sSh[tid] = stats2[32 + tid]; }
  __syncthreads();
  const int n = blockIdx.x * 256 + tid;
  if (n >= NPTS) return;
  float o[8] = {0,0,0,0,0,0,0,0};
  const float4* src = (const float4*)(h2 + (size_t)n * C_S);
#pragma unroll
  for (int q = 0; q < 8; ++q) {
    const float4 f = src[q];
    const float av[4] = { f.x, f.y, f.z, f.w };
#pragma unroll
    for (int u = 0; u < 4; ++u) {
      const int c = 4 * q + u;
      const float a = av[u] * sSc[c] + sSh[c];
#pragma unroll
      for (int j = 0; j < C_OUT; ++j) o[j] += a * sW3[c * C_OUT + j];
    }
  }
  float4* dst = (float4*)(out + (size_t)n * C_OUT);
  dst[0] = make_float4(o[0], o[1], o[2], o[3]);
  dst[1] = make_float4(o[4], o[5], o[6], o[7]);
}

// ---------------------------------------------------------------------------
extern "C" void kernel_launch(void* const* d_in, const int* in_sizes, int n_in,
                              void* d_out, int out_size, void* d_ws, size_t ws_size,
                              hipStream_t stream) {
  const float*         x        = (const float*)d_in[0];
  const int*           nbr_idx  = (const int*)d_in[1];
  const unsigned char* nbr_mask = (const unsigned char*)d_in[2];   // jax bool = 1 byte
  const float*         W1       = (const float*)d_in[3];
  const float*         gamma1   = (const float*)d_in[4];
  const float*         beta1    = (const float*)d_in[5];
  const float*         W2       = (const float*)d_in[6];
  const float*         gamma2   = (const float*)d_in[7];
  const float*         beta2    = (const float*)d_in[8];
  const float*         W3       = (const float*)d_in[9];
  float*               out      = (float*)d_out;

  // workspace carve-up (all segments 256B aligned)
  char* ws = (char*)d_ws;
  __bf16* W1b   = (__bf16*)ws;  ws += (size_t)W1B_ELEMS * 2;           // 1.77 MB
  float*  h1    = (float*)ws;   ws += (size_t)NPTS * C_MID * 4;        // 153.6 MB
  float*  part1 = (float*)ws;   ws += (size_t)NWG1 * 256 * 4;          // 2.4 MB
  float*  stats1= (float*)ws;   ws += 256 * 4;
  float*  h2    = (float*)ws;   ws += (size_t)NPTS * C_S * 4;          // 38.4 MB
  float*  part2 = (float*)ws;   ws += (size_t)NWG2 * 64 * 4;
  float*  stats2= (float*)ws;   ws += 64 * 4;
  (void)ws_size; (void)in_sizes; (void)n_in; (void)out_size;

  k_pack_w1<<<W1B_ELEMS / 256, 256, 0, stream>>>(W1, W1b);
  k_conv   <<<NWG1, 256, 0, stream>>>(x, nbr_idx, nbr_mask, W1b, h1, part1);
  k_reduce1<<<1, 256, 0, stream>>>(part1, gamma1, beta1, stats1);
  k_stage2 <<<NWG2, 256, 0, stream>>>(h1, stats1, W2, h2, part2);
  k_reduce2<<<1, 64, 0, stream>>>(part2, gamma2, beta2, stats2);
  k_stage3 <<<(NPTS + 255) / 256, 256, 0, stream>>>(h2, stats2, W3, out);
}
